// Qwen3Model_49168785605243
// MI455X (gfx1250) — compile-verified, matrix operand order checked by reading, not run on hardware
//
#include <hip/hip_runtime.h>
#include <hip/hip_bf16.h>

typedef __bf16 bf16_t;
typedef bf16_t bf16x16 __attribute__((ext_vector_type(16)));
typedef bf16_t bf16x8  __attribute__((ext_vector_type(8)));
typedef float  fx8     __attribute__((ext_vector_type(8)));

// Model dims (fixed by the reference)
constexpr int Lc = 2, Dc = 2048, Hc = 16, KVc = 8, HDc = 128, FFc = 6144, Vc = 32000, Tc = 2048;
constexpr float EPSc = 1e-6f;
constexpr float NEGc = -1e30f;
constexpr float THETAc = 1000000.0f;

// Native fp32 -> bf16 convert (v_cvt_pk_bf16_f32 per pair)
__device__ __forceinline__ bf16_t tobf(float f) { return (bf16_t)f; }

__device__ __forceinline__ unsigned packbf2(float a, float b) {
#if __has_builtin(__builtin_amdgcn_cvt_pk_bf16_f32)
  typedef bf16_t bf16x2 __attribute__((ext_vector_type(2)));
  bf16x2 v = __builtin_amdgcn_cvt_pk_bf16_f32(a, b);
  return __builtin_bit_cast(unsigned, v);
#else
  unsigned short ua = __builtin_bit_cast(unsigned short, (bf16_t)a);
  unsigned short ub = __builtin_bit_cast(unsigned short, (bf16_t)b);
  return (unsigned)ua | ((unsigned)ub << 16);
#endif
}

// D = A(16x32 bf16) * B(32x16 bf16) + C (fp32), wave32 WMMA
__device__ __forceinline__ fx8 wmma_bf16(bf16x16 a, bf16x16 b, fx8 c) {
  return __builtin_amdgcn_wmma_f32_16x16x32_bf16(false, a, false, b, (short)0, c, false, false);
}

// 16-bit fragment = two contiguous 8-element runs per lane: [kb..kb+7], [kb+16..kb+23].
__device__ __forceinline__ bf16x16 ld_frag(const bf16_t* base) {
  bf16x8 lo = *(const bf16x8*)(base);
  bf16x8 hi = *(const bf16x8*)(base + 16);
  return __builtin_shufflevector(lo, hi, 0, 1, 2, 3, 4, 5, 6, 7, 8, 9, 10, 11, 12, 13, 14, 15);
}

__device__ __forceinline__ int kmap(int e) { return e < 8 ? e : e + 8; }

// Async 16B global->LDS copy (GLOBAL_LOAD_ASYNC_TO_LDS_B128, tracked by ASYNCcnt)
__device__ __forceinline__ void async_copy16(unsigned lds_off, const void* gaddr) {
  asm volatile("global_load_async_to_lds_b128 %0, %1, off"
               :
               : "v"(lds_off), "v"((unsigned long long)(uintptr_t)gaddr)
               : "memory");
}
__device__ __forceinline__ void wait_async0() {
  asm volatile("s_wait_asynccnt 0x0" ::: "memory");
}

// ---------------------------------------------------------------------------
// Tiled GEMM: C[M,N] (+)= A[M,K](bf16) x B  (fp32 [K,N], or TRANSB: [N,K])
// 256 threads, BM=BN=128, BK=64, 8 waves each computing a 32x64 region.
// A tile staged with async LDS DMA (already bf16); B tile converted fp32->bf16
// with packed cvt and stored TRANSPOSED so both fragments read contiguously.
// ---------------------------------------------------------------------------
template <bool TRANSB, bool ACCUM>
__global__ __launch_bounds__(256) void gemm128(const bf16_t* __restrict__ A,
                                               const float* __restrict__ B,
                                               float* __restrict__ C,
                                               int M, int N, int K) {
  constexpr int BM = 128, BN = 128, BK = 64;
  constexpr int LDA = BK + 8;   // 72 elems = 144B row stride (16B aligned)
  __shared__ bf16_t As[BM][LDA];        // [m][k]
  __shared__ bf16_t Bst[BN][LDA];       // [n][k]  (transposed tile)

  const int tid  = threadIdx.x;
  const int m0   = blockIdx.y * BM;
  const int n0   = blockIdx.x * BN;
  const int lane = tid & 31;
  const int wave = tid >> 5;
  const int wm   = wave >> 1;        // 0..3 -> row block of 32
  const int wn   = wave & 1;         // 0..1 -> col block of 64
  const int row16 = lane & 15;
  const int kb    = (lane >> 4) << 3;
  const int grp8  = (lane >> 4) << 3;

  fx8 acc[2][4];
#pragma unroll
  for (int i = 0; i < 2; i++)
#pragma unroll
    for (int j = 0; j < 4; j++)
#pragma unroll
      for (int r = 0; r < 8; r++) acc[i][j][r] = 0.0f;

  for (int k0 = 0; k0 < K; k0 += BK) {
    // --- A tile: async LDS DMA, 16B chunks (bf16 source, no conversion) ---
#pragma unroll
    for (int i = 0; i < 4; i++) {
      int cidx = tid + i * 256;           // 1024 chunks of 8 bf16
      int r    = cidx >> 3;               // 8 chunks per 64-wide row
      int c8   = (cidx & 7) << 3;         // element offset, 16B aligned
      unsigned lds_off = (unsigned)(uintptr_t)&As[r][c8];
      async_copy16(lds_off, &A[(size_t)(m0 + r) * K + (k0 + c8)]);
    }
    // --- B tile transposed: fp32 -> bf16 packed converts ---
#pragma unroll
    for (int i = 0; i < 16; i++) {
      int idx = tid + i * 256;      // 4096 pairs
      int c   = idx & 127;          // n within tile
      int r0  = (idx >> 7) << 1;    // even k within tile
      float b0v, b1v;
      if (TRANSB) {
        float2 b2 = *(const float2*)&B[(size_t)(n0 + c) * K + (k0 + r0)];
        b0v = b2.x; b1v = b2.y;
      } else {
        b0v = B[(size_t)(k0 + r0)     * N + (n0 + c)];
        b1v = B[(size_t)(k0 + r0 + 1) * N + (n0 + c)];
      }
      *(unsigned*)&Bst[c][r0] = packbf2(b0v, b1v);
    }
    // Prefetch next K tile (global_prefetch_b8)
    if (k0 + BK < K) {
      __builtin_prefetch(&A[(size_t)(m0 + (tid >> 1)) * K + (k0 + BK)], 0, 3);
      if (TRANSB)
        __builtin_prefetch(&B[(size_t)(n0 + (tid >> 1)) * K + (k0 + BK)], 0, 3);
      else
        __builtin_prefetch(&B[(size_t)(k0 + BK + (tid >> 3)) * N + n0], 0, 3);
    }
    wait_async0();
    __syncthreads();

#pragma unroll
    for (int kc = 0; kc < 2; kc++) {
#pragma unroll
      for (int mi = 0; mi < 2; mi++) {
        bf16x16 af = ld_frag(&As[wm * 32 + mi * 16 + row16][kc * 32 + kb]);
#pragma unroll
        for (int ni = 0; ni < 4; ni++) {
          bf16x16 bfr = ld_frag(&Bst[wn * 64 + ni * 16 + row16][kc * 32 + kb]);
          acc[mi][ni] = wmma_bf16(af, bfr, acc[mi][ni]);
        }
      }
    }
    __syncthreads();
  }

#pragma unroll
  for (int mi = 0; mi < 2; mi++)
#pragma unroll
    for (int ni = 0; ni < 4; ni++)
#pragma unroll
      for (int r = 0; r < 8; r++) {
        int gm = m0 + wm * 32 + mi * 16 + r + grp8;
        int gn = n0 + wn * 64 + ni * 16 + row16;
        size_t off = (size_t)gm * N + gn;
        if (ACCUM) C[off] += acc[mi][ni][r];
        else       C[off]  = acc[mi][ni][r];
      }
}

// ---------------------------------------------------------------------------
// RMSNorm over ncol (even) features; one block per row; bf16 output.
// ---------------------------------------------------------------------------
__global__ __launch_bounds__(256) void rmsnorm_k(const float* __restrict__ x,
                                                 const float* __restrict__ g,
                                                 bf16_t* __restrict__ o, int ncol) {
  __shared__ float red[256];
  int row = blockIdx.x;
  const float* xr = x + (size_t)row * ncol;
  float ss = 0.0f;
  for (int c = threadIdx.x * 2; c < ncol; c += 512) {
    float2 v = *(const float2*)&xr[c];
    ss += v.x * v.x + v.y * v.y;
  }
  red[threadIdx.x] = ss;
  __syncthreads();
  for (int s = 128; s > 0; s >>= 1) {
    if (threadIdx.x < (unsigned)s) red[threadIdx.x] += red[threadIdx.x + s];
    __syncthreads();
  }
  float inv = rsqrtf(red[0] / (float)ncol + EPSc);
  bf16_t* orow = o + (size_t)row * ncol;
  for (int c = threadIdx.x * 2; c < ncol; c += 512) {
    float2 v = *(const float2*)&xr[c];
    float2 gg = *(const float2*)&g[c];
    *(unsigned*)&orow[c] = packbf2(v.x * inv * gg.x, v.y * inv * gg.y);
  }
}

// ---------------------------------------------------------------------------
// Per-head RMSNorm (HD=128) + RoPE, in place (fp32). One wave per (token, head).
// ---------------------------------------------------------------------------
__global__ __launch_bounds__(128) void qknorm_rope_k(float* __restrict__ qk,
                                                     const float* __restrict__ nsc,
                                                     int nheads) {
  int w    = blockIdx.x * 4 + (threadIdx.x >> 5);
  int lane = threadIdx.x & 31;
  int t    = w / nheads;
  int head = w % nheads;
  float* p = qk + (size_t)t * nheads * HDc + (size_t)head * HDc;

  float x0 = p[lane], x1 = p[lane + 32], x2 = p[lane + 64], x3 = p[lane + 96];
  float ss = x0 * x0 + x1 * x1 + x2 * x2 + x3 * x3;
#pragma unroll
  for (int m = 1; m < 32; m <<= 1) ss += __shfl_xor(ss, m, 32);
  float inv = rsqrtf(ss / (float)HDc + EPSc);
  x0 *= inv * nsc[lane];       x1 *= inv * nsc[lane + 32];
  x2 *= inv * nsc[lane + 64];  x3 *= inv * nsc[lane + 96];

  float f0 = __powf(THETAc, -((2.0f * lane) / (float)HDc));
  float f1 = __powf(THETAc, -((2.0f * (lane + 32)) / (float)HDc));
  float a0 = (float)t * f0, a1 = (float)t * f1;
  float s0, c0, s1, c1;
  __sincosf(a0, &s0, &c0);
  __sincosf(a1, &s1, &c1);

  p[lane]      = x0 * c0 - x2 * s0;
  p[lane + 64] = x2 * c0 + x0 * s0;
  p[lane + 32] = x1 * c1 - x3 * s1;
  p[lane + 96] = x3 * c1 + x1 * s1;
}

// ---------------------------------------------------------------------------
// Flash attention: one wave per (head, 16-query tile), GQA head->kv = head>>1.
// Output written directly in bf16 (consumed only by the Wo GEMM).
// ---------------------------------------------------------------------------
__global__ __launch_bounds__(128) void attn_k(const float* __restrict__ q,
                                              const float* __restrict__ kk,
                                              const float* __restrict__ vv,
                                              bf16_t* __restrict__ o) {
  __shared__ bf16_t plds[4][16][40];   // 80B row stride -> 16B aligned frag reads
  const int wv   = threadIdx.x >> 5;
  const int lane = threadIdx.x & 31;
  const int w    = blockIdx.x * 4 + wv;
  const int head = w >> 7;               // T/16 = 128 qtiles per head
  const int qt   = (w & 127) << 4;
  const int kvh  = head >> 1;            // G = H/KV = 2
  const int row16 = lane & 15;
  const int grp8  = (lane >> 4) << 3;
  const int kb    = grp8;
  const float scale = 0.088388347648318447f; // 1/sqrt(128)

  bf16x16 qf[4];
#pragma unroll
  for (int c = 0; c < 4; c++)
#pragma unroll
    for (int e = 0; e < 16; e++)
      qf[c][e] = tobf(q[(size_t)(qt + row16) * (Hc * HDc) + head * HDc + c * 32 + kb + kmap(e)]);

  float m[8], l[8];
  fx8 oacc[8];
#pragma unroll
  for (int r = 0; r < 8; r++) { m[r] = -3e38f; l[r] = 0.0f; }
#pragma unroll
  for (int t = 0; t < 8; t++)
#pragma unroll
    for (int r = 0; r < 8; r++) oacc[t][r] = 0.0f;

  for (int s0 = 0; s0 < qt + 16; s0 += 32) {
    fx8 s[2];
#pragma unroll
    for (int j = 0; j < 2; j++)
#pragma unroll
      for (int r = 0; r < 8; r++) s[j][r] = 0.0f;

#pragma unroll
    for (int c = 0; c < 4; c++) {
      bf16x16 b0, b1;
#pragma unroll
      for (int e = 0; e < 16; e++) {
        int kfe = c * 32 + kb + kmap(e);
        b0[e] = tobf(kk[(size_t)(s0 + row16)      * (KVc * HDc) + kvh * HDc + kfe]);
        b1[e] = tobf(kk[(size_t)(s0 + 16 + row16) * (KVc * HDc) + kvh * HDc + kfe]);
      }
      s[0] = wmma_bf16(qf[c], b0, s[0]);
      s[1] = wmma_bf16(qf[c], b1, s[1]);
    }

    // scale + causal mask
#pragma unroll
    for (int j = 0; j < 2; j++) {
      int scol = s0 + j * 16 + row16;
#pragma unroll
      for (int r = 0; r < 8; r++) {
        int qrow = qt + r + grp8;
        float v = s[j][r] * scale;
        s[j][r] = (scol > qrow) ? NEGc : v;
      }
    }

    // online softmax (rows spread across 16 lanes at fixed vgpr index)
#pragma unroll
    for (int r = 0; r < 8; r++) {
      float mx = fmaxf(s[0][r], s[1][r]);
#pragma unroll
      for (int mm = 1; mm < 16; mm <<= 1) mx = fmaxf(mx, __shfl_xor(mx, mm, 16));
      float mnew = fmaxf(m[r], mx);
      float corr = __expf(m[r] - mnew);
      float p0 = __expf(s[0][r] - mnew);
      float p1 = __expf(s[1][r] - mnew);
      s[0][r] = p0; s[1][r] = p1;
      float ps = p0 + p1;
#pragma unroll
      for (int mm = 1; mm < 16; mm <<= 1) ps += __shfl_xor(ps, mm, 16);
      l[r] = l[r] * corr + ps;
      m[r] = mnew;
#pragma unroll
      for (int t = 0; t < 8; t++) oacc[t][r] *= corr;
    }

    // transpose P (C layout) -> A-fragment layout through wave-private LDS
#pragma unroll
    for (int j = 0; j < 2; j++)
#pragma unroll
      for (int r = 0; r < 8; r++)
        plds[wv][r + grp8][j * 16 + row16] = tobf(s[j][r]);
    bf16x16 pf = ld_frag(&plds[wv][row16][kb]);

    // O += P(16x32) * V(32x128)
#pragma unroll
    for (int c2 = 0; c2 < 8; c2++) {
      bf16x16 vb;
#pragma unroll
      for (int e = 0; e < 16; e++)
        vb[e] = tobf(vv[(size_t)(s0 + kb + kmap(e)) * (KVc * HDc) + kvh * HDc + c2 * 16 + row16]);
      oacc[c2] = wmma_bf16(pf, vb, oacc[c2]);
    }
  }

#pragma unroll
  for (int c2 = 0; c2 < 8; c2++)
#pragma unroll
    for (int r = 0; r < 8; r++)
      o[(size_t)(qt + r + grp8) * (Hc * HDc) + head * HDc + c2 * 16 + row16] =
          tobf(oacc[c2][r] / l[r]);
}

// ---------------------------------------------------------------------------
// ub_bf16 = silu(u) * g   (2 elements per thread, packed bf16 store)
// ---------------------------------------------------------------------------
__global__ __launch_bounds__(256) void silu_mul_k(const float* __restrict__ u,
                                                  const float* __restrict__ g,
                                                  bf16_t* __restrict__ ob,
                                                  size_t npair) {
  size_t i = (size_t)blockIdx.x * 256 + threadIdx.x;
  if (i < npair) {
    float2 a = *(const float2*)&u[i * 2];
    float2 b = *(const float2*)&g[i * 2];
    float s0 = a.x / (1.0f + __expf(-a.x)) * b.x;
    float s1 = a.y / (1.0f + __expf(-a.y)) * b.y;
    *(unsigned*)&ob[i * 2] = packbf2(s0, s1);
  }
}

// ---------------------------------------------------------------------------
// x[t,:] = emb[ids[t],:]
// ---------------------------------------------------------------------------
__global__ __launch_bounds__(256) void embed_k(const int* __restrict__ ids,
                                               const float* __restrict__ emb,
                                               float* __restrict__ x) {
  int t = blockIdx.x;
  int id = ids[t];
  const float* e = emb + (size_t)id * Dc;
  float* xr = x + (size_t)t * Dc;
  for (int c = threadIdx.x; c < Dc; c += 256) xr[c] = e[c];
}

// ---------------------------------------------------------------------------
extern "C" void kernel_launch(void* const* d_in, const int* in_sizes, int n_in,
                              void* d_out, int out_size, void* d_ws, size_t ws_size,
                              hipStream_t stream) {
  (void)in_sizes; (void)n_in; (void)out_size; (void)ws_size;
  const int*   ids = (const int*)d_in[0];
  const float* emb = (const float*)d_in[1];
  const float* Wq  = (const float*)d_in[2];
  const float* Wk  = (const float*)d_in[3];
  const float* Wv  = (const float*)d_in[4];
  const float* Wo  = (const float*)d_in[5];
  const float* qn  = (const float*)d_in[6];
  const float* kn  = (const float*)d_in[7];
  const float* W1  = (const float*)d_in[8];
  const float* W2  = (const float*)d_in[9];
  const float* W3  = (const float*)d_in[10];
  const float* ln1 = (const float*)d_in[11];
  const float* ln2 = (const float*)d_in[12];
  const float* lnf = (const float*)d_in[13];
  float* out = (float*)d_out;

  const size_t TD = (size_t)Tc * Dc;
  float*  x    = (float*)d_ws;                       // T x D fp32
  float*  qb   = x + TD;                             // T x H*HD fp32
  float*  kbuf = qb + TD;                            // T x KV*HD fp32
  float*  vbuf = kbuf + TD / 2;                      // T x KV*HD fp32
  float*  ub   = vbuf + TD / 2;                      // T x FF fp32
  float*  gb   = ub + (size_t)Tc * FFc;              // T x FF fp32
  bf16_t* hb   = (bf16_t*)(gb + (size_t)Tc * FFc);   // T x D bf16
  bf16_t* attb = hb + TD;                            // T x H*HD bf16
  bf16_t* ubb  = attb + TD;                          // T x FF bf16

  dim3 b256(256), b128(128);
  embed_k<<<Tc, b256, 0, stream>>>(ids, emb, x);

  for (int l = 0; l < Lc; l++) {
    rmsnorm_k<<<Tc, b256, 0, stream>>>(x, ln1 + (size_t)l * Dc, hb, Dc);
    gemm128<false, false><<<dim3((Hc * HDc) / 128, Tc / 128), b256, 0, stream>>>(
        hb, Wq + (size_t)l * Dc * Hc * HDc, qb, Tc, Hc * HDc, Dc);
    gemm128<false, false><<<dim3((KVc * HDc) / 128, Tc / 128), b256, 0, stream>>>(
        hb, Wk + (size_t)l * Dc * KVc * HDc, kbuf, Tc, KVc * HDc, Dc);
    gemm128<false, false><<<dim3((KVc * HDc) / 128, Tc / 128), b256, 0, stream>>>(
        hb, Wv + (size_t)l * Dc * KVc * HDc, vbuf, Tc, KVc * HDc, Dc);
    qknorm_rope_k<<<(Tc * Hc) / 4, b128, 0, stream>>>(qb, qn + (size_t)l * HDc, Hc);
    qknorm_rope_k<<<(Tc * KVc) / 4, b128, 0, stream>>>(kbuf, kn + (size_t)l * HDc, KVc);
    attn_k<<<(Hc * (Tc / 16)) / 4, b128, 0, stream>>>(qb, kbuf, vbuf, attb);
    gemm128<false, true><<<dim3(Dc / 128, Tc / 128), b256, 0, stream>>>(
        attb, Wo + (size_t)l * Hc * HDc * Dc, x, Tc, Dc, Hc * HDc);
    rmsnorm_k<<<Tc, b256, 0, stream>>>(x, ln2 + (size_t)l * Dc, hb, Dc);
    gemm128<false, false><<<dim3(FFc / 128, Tc / 128), b256, 0, stream>>>(
        hb, W1 + (size_t)l * Dc * FFc, ub, Tc, FFc, Dc);
    gemm128<false, false><<<dim3(FFc / 128, Tc / 128), b256, 0, stream>>>(
        hb, W2 + (size_t)l * Dc * FFc, gb, Tc, FFc, Dc);
    size_t npair = (size_t)Tc * FFc / 2;
    silu_mul_k<<<(unsigned)((npair + 255) / 256), b256, 0, stream>>>(ub, gb, ubb, npair);
    gemm128<false, true><<<dim3(Dc / 128, Tc / 128), b256, 0, stream>>>(
        ubb, W3 + (size_t)l * FFc * Dc, x, Tc, Dc, FFc);
  }

  rmsnorm_k<<<Tc, b256, 0, stream>>>(x, lnf, hb, Dc);
  gemm128<true, false><<<dim3(Vc / 128, Tc / 128), b256, 0, stream>>>(
      hb, emb, out, Tc, Vc, Dc);
}
